// AutoencoderRNN_VAE_v2_20100446946152
// MI455X (gfx1250) — matrix operand order repaired and checked by reading, not compiled, not run on hardware
//
#include <hip/hip_runtime.h>
#include <hip/hip_bf16.h>
#include <math.h>

// ---------------------------------------------------------------------------
// AutoencoderRNN VAE forward for MI455X (gfx1250), wave32 + WMMA f16->f32.
//
// All dense math runs through one generic WMMA GEMM kernel:
//   out = act( A1 @ B1^T + A2 @ B2^T + C0 + bias ), out -> f32 and/or f16
// A operands are pre-packed f16 row-major with K padded to a multiple of 32;
// weights are pre-converted f16 with the padded-K region zero-filled, so no
// K or M guards are needed: every fragment load is an unconditional b128.
// ---------------------------------------------------------------------------

typedef __attribute__((ext_vector_type(16))) _Float16 v16h;
typedef __attribute__((ext_vector_type(8)))  _Float16 v8h;
typedef __attribute__((ext_vector_type(8)))  float    v8f;

#define BM 128           // block tile rows  (all M are multiples of 128)
#define BN 128           // block tile cols
// 8 waves per block: 4 along M x 2 along N; wave tile = 32(M) x 64(N)

// ---- fragment loaders (CDNA5 16-bit WMMA VGPR layouts, ISA 7.12.2) --------

// A: 16x32 (MxK), f16 row-major (ld = Kp). lane<16: K = 0..7 & 16..23,
// lanes 16..31: K = 8..15 & 24..31.  Two unconditional 16B loads.
__device__ __forceinline__ v16h load_a_frag(const _Float16* __restrict__ A,
                                            int ldap, int row0, int kb, int lane) {
    const int row   = row0 + (lane & 15);
    const int khalf = (lane >> 4) << 3;          // 0 or 8
    const _Float16* p = A + (size_t)row * ldap + kb + khalf;
    const v8h lo = *(const v8h*)(p);
    const v8h hi = *(const v8h*)(p + 16);
    v16h a;
#pragma unroll
    for (int j = 0; j < 8; ++j) { a[j] = lo[j]; a[8 + j] = hi[j]; }
    return a;
}

// B: 32x16 (KxN), weight stored row-major (N, Kp) f16, padded-K zero-filled.
// lane<16 -> K 0..7 & 16..23, lane>=16 -> K 8..15 & 24..31. Branchless column clamp.
__device__ __forceinline__ v16h load_b_frag(const _Float16* __restrict__ Bw,
                                            int Kp, int N, int col0,
                                            int kb, int lane) {
    int col = col0 + (lane & 15);
    if (col >= N) col = N - 1;                   // clamp: masked at store time
    const int ksel = (lane >> 4) << 3;           // 0 or 8
    const _Float16* p = Bw + (size_t)col * Kp + kb + ksel;
    const v8h lo = *(const v8h*)(p);
    const v8h hi = *(const v8h*)(p + 16);
    v16h b;
#pragma unroll
    for (int j = 0; j < 8; ++j) { b[j] = lo[j]; b[8 + j] = hi[j]; }
    return b;
}

// ---- main GEMM kernel ------------------------------------------------------

__global__ __launch_bounds__(256)
void wmma_gemm_kernel(const _Float16* __restrict__ A1, int lda1,
                      const _Float16* __restrict__ B1, int K1p,
                      const _Float16* __restrict__ A2, int lda2,
                      const _Float16* __restrict__ B2, int K2p,
                      const float* __restrict__ C0, int ldc0,
                      const float* __restrict__ bias,
                      float* __restrict__ C, int ldc,
                      _Float16* __restrict__ C16, int ldc16,
                      int M, int N, int act)
{
    const int tid   = threadIdx.x;
    const int lane  = tid & 31;
    const int wid   = tid >> 5;
    const int waveM = wid & 3;                   // 0..3
    const int waveN = wid >> 2;                  // 0..1
    const int m0 = blockIdx.y * BM + waveM * 32;
    const int n0 = blockIdx.x * BN + waveN * 64;
    (void)M;

    v8f acc[2][4];
#pragma unroll
    for (int mi = 0; mi < 2; ++mi)
#pragma unroll
        for (int ni = 0; ni < 4; ++ni)
#pragma unroll
            for (int r = 0; r < 8; ++r) acc[mi][ni][r] = 0.f;

    // ---- pair 1 ----
    for (int kb = 0; kb < K1p; kb += 32) {
        const v16h a0 = load_a_frag(A1, lda1, m0,      kb, lane);
        const v16h a1 = load_a_frag(A1, lda1, m0 + 16, kb, lane);
#pragma unroll
        for (int ni = 0; ni < 4; ++ni) {
            const v16h b = load_b_frag(B1, K1p, N, n0 + ni * 16, kb, lane);
            acc[0][ni] = __builtin_amdgcn_wmma_f32_16x16x32_f16(
                false, a0, false, b, (short)0, acc[0][ni], false, false);
            acc[1][ni] = __builtin_amdgcn_wmma_f32_16x16x32_f16(
                false, a1, false, b, (short)0, acc[1][ni], false, false);
        }
    }

    // ---- pair 2 (optional) ----
    if (A2 != nullptr) {
        for (int kb = 0; kb < K2p; kb += 32) {
            const v16h a0 = load_a_frag(A2, lda2, m0,      kb, lane);
            const v16h a1 = load_a_frag(A2, lda2, m0 + 16, kb, lane);
#pragma unroll
            for (int ni = 0; ni < 4; ++ni) {
                const v16h b = load_b_frag(B2, K2p, N, n0 + ni * 16, kb, lane);
                acc[0][ni] = __builtin_amdgcn_wmma_f32_16x16x32_f16(
                    false, a0, false, b, (short)0, acc[0][ni], false, false);
                acc[1][ni] = __builtin_amdgcn_wmma_f32_16x16x32_f16(
                    false, a1, false, b, (short)0, acc[1][ni], false, false);
            }
        }
    }

    // ---- epilogue: C layout (ISA 7.12.2): vgpr r, lane l ->
    //      row = r + 8*(l>>4), col = l&15 within the 16x16 tile ----
#pragma unroll
    for (int mi = 0; mi < 2; ++mi) {
#pragma unroll
        for (int ni = 0; ni < 4; ++ni) {
            const int col = n0 + ni * 16 + (lane & 15);
            const bool cok = (col < N);
            const float bv = (bias != nullptr && cok) ? bias[col] : 0.f;
#pragma unroll
            for (int r = 0; r < 8; ++r) {
                const int row = m0 + mi * 16 + r + ((lane >> 4) << 3);
                float v = acc[mi][ni][r] + bv;
                if (C0 != nullptr && cok) v += C0[(size_t)row * ldc0 + col];
                if (act == 1) v = tanhf(v);
                if (cok) {
                    if (C   != nullptr) C[(size_t)row * ldc + col] = v;
                    if (C16 != nullptr) C16[(size_t)row * ldc16 + col] = (_Float16)v;
                }
            }
        }
    }
}

// ---- LSTM cell pointwise kernel (gate order i,f,g,o); h emitted as f16 ----

__global__ void lstm_cell_kernel(const float* __restrict__ gates,
                                 float* __restrict__ c,
                                 _Float16* __restrict__ h16,
                                 int Beff, int Hdim)
{
    const int idx = blockIdx.x * blockDim.x + threadIdx.x;
    if (idx >= Beff * Hdim) return;
    const int b = idx / Hdim;
    const int j = idx - b * Hdim;
    const float* g = gates + (size_t)b * 4 * Hdim;
    const float gi = 1.f / (1.f + __expf(-g[j]));
    const float gf = 1.f / (1.f + __expf(-g[Hdim + j]));
    const float gg = tanhf(g[2 * Hdim + j]);
    const float go = 1.f / (1.f + __expf(-g[3 * Hdim + j]));
    const float cn = gf * c[idx] + gi * gg;
    c[idx]   = cn;
    h16[idx] = (_Float16)(go * tanhf(cn));
}

// ---- f32 -> f16 pack/convert (weights AND activations), K zero-padded -----

__global__ void cvt_pack_kernel(const float* __restrict__ src,
                                _Float16* __restrict__ dst,
                                int Nrows, int K, int Kp, int src_ld, int col_off)
{
    const int idx = blockIdx.x * blockDim.x + threadIdx.x;
    if (idx >= Nrows * Kp) return;
    const int n = idx / Kp;
    const int k = idx - n * Kp;
    dst[idx] = (k < K) ? (_Float16)src[(size_t)n * src_ld + col_off + k]
                       : (_Float16)0.f;
}

__global__ void addvec_kernel(const float* __restrict__ a,
                              const float* __restrict__ b,
                              float* __restrict__ o, int n)
{
    const int i = blockIdx.x * blockDim.x + threadIdx.x;
    if (i < n) o[i] = a[i] + b[i];
}

__global__ void fill0_f32_kernel(float* __restrict__ p, size_t n)
{
    const size_t i = (size_t)blockIdx.x * blockDim.x + threadIdx.x;
    if (i < n) p[i] = 0.f;
}

__global__ void fill0_f16_kernel(_Float16* __restrict__ p, size_t n)
{
    const size_t i = (size_t)blockIdx.x * blockDim.x + threadIdx.x;
    if (i < n) p[i] = (_Float16)0.f;
}

// ---------------------------------------------------------------------------

extern "C" void kernel_launch(void* const* d_in, const int* in_sizes, int n_in,
                              void* d_out, int out_size, void* d_ws, size_t ws_size,
                              hipStream_t stream)
{
    (void)in_sizes; (void)n_in; (void)out_size; (void)ws_size;

    // ---- problem constants ----
    const int Bb = 128, T = 750, Wt = 50, F = 25, H = 512, Z = 256, Mm = 512, C3D = 58;
    const int Beff = Bb * T / Wt;            // 1920 = 15 * 128
    const int G    = 4 * H;                  // 2048
    const int Fp   = 32;                     // F padded to 32 for WMMA K
    const int CH   = 6400;                   // 50*128; 96000/6400 = 15 exact chunks

    // ---- inputs (setup_inputs dict order) ----
    const float* le      = (const float*)d_in[0];   // (B,T,F)
    /* d_in[1] listener_3dmm is unused by the reference computation */
    const float* enc_Wih = (const float*)d_in[2];
    const float* enc_Whh = (const float*)d_in[3];
    const float* enc_bih = (const float*)d_in[4];
    const float* enc_bhh = (const float*)d_in[5];
    const float* mu_W    = (const float*)d_in[6];
    const float* mu_b    = (const float*)d_in[7];
    const float* lv_W    = (const float*)d_in[8];
    const float* lv_b    = (const float*)d_in[9];
    const float* zdec_W  = (const float*)d_in[10];
    const float* zdec_b  = (const float*)d_in[11];
    const float* dec_Wih = (const float*)d_in[12];  // (2048, 537)
    const float* dec_Whh = (const float*)d_in[13];
    const float* dec_bih = (const float*)d_in[14];
    const float* dec_bhh = (const float*)d_in[15];
    const float* mlp1_W  = (const float*)d_in[16];
    const float* mlp1_b  = (const float*)d_in[17];
    const float* mlp2_W  = (const float*)d_in[18];
    const float* mlp2_b  = (const float*)d_in[19];
    const float* out_W   = (const float*)d_in[20];
    const float* out_b   = (const float*)d_in[21];
    const float* cr1_W   = (const float*)d_in[22];
    const float* cr1_b   = (const float*)d_in[23];
    const float* cr2_W   = (const float*)d_in[24];
    const float* cr2_b   = (const float*)d_in[25];

    // ---- outputs: pred, coeff_3dmm, mu, logvar concatenated flat ----
    float* out    = (float*)d_out;
    float* pred   = out;                                        // 128*750*25
    float* coeff  = out + (size_t)Bb * T * F;                   // 128*750*58
    float* mu_out = coeff + (size_t)Bb * T * C3D;               // 1920*256
    float* lv_out = mu_out + (size_t)Beff * Z;                  // 1920*256

    // ---- workspace cursor allocator ----
    char* wsb = (char*)d_ws;
    size_t cur = 0;
    auto alloc = [&](size_t bytes) -> void* {
        cur = (cur + 255) & ~(size_t)255;
        void* p = (void*)(wsb + cur);
        cur += bytes;
        return p;
    };

    // f16 weights (K padded to multiple of 32, zero-filled)
    _Float16* encWih16  = (_Float16*)alloc((size_t)G * Fp * 2);
    _Float16* encWhh16  = (_Float16*)alloc((size_t)G * H  * 2);
    _Float16* muW16     = (_Float16*)alloc((size_t)Z * H  * 2);
    _Float16* lvW16     = (_Float16*)alloc((size_t)Z * H  * 2);
    _Float16* zdec16    = (_Float16*)alloc((size_t)H * Z  * 2);
    _Float16* decWihH16 = (_Float16*)alloc((size_t)G * H  * 2);
    _Float16* decWihY16 = (_Float16*)alloc((size_t)G * Fp * 2);
    _Float16* decWhh16  = (_Float16*)alloc((size_t)G * H  * 2);
    _Float16* mlp116    = (_Float16*)alloc((size_t)Mm * H  * 2);
    _Float16* mlp216    = (_Float16*)alloc((size_t)Mm * Mm * 2);
    _Float16* outW16    = (_Float16*)alloc((size_t)F * Mm * 2);
    _Float16* cr116     = (_Float16*)alloc((size_t)H * Fp * 2);
    _Float16* cr216     = (_Float16*)alloc((size_t)C3D * H * 2);

    // f16 packed activations
    _Float16* xpack     = (_Float16*)alloc((size_t)Wt * Beff * Fp * 2); // per-t slabs
    _Float16* hpackE    = (_Float16*)alloc((size_t)Beff * H * 2);
    _Float16* hpackD    = (_Float16*)alloc((size_t)Beff * H * 2);
    _Float16* mupack    = (_Float16*)alloc((size_t)Beff * Z * 2);
    _Float16* hypack    = (_Float16*)alloc((size_t)Beff * H * 2);
    _Float16* m1pack    = (_Float16*)alloc((size_t)Beff * Mm * 2);
    _Float16* m2pack    = (_Float16*)alloc((size_t)Beff * Mm * 2);
    _Float16* ypack     = (_Float16*)alloc((size_t)Beff * Fp * 2);
    _Float16* predpack  = (_Float16*)alloc((size_t)Bb * T * Fp * 2);
    _Float16* tmpcpack  = (_Float16*)alloc((size_t)CH * H * 2);

    // f32 scratch
    float* c_enc  = (float*)alloc((size_t)Beff * H * 4);
    float* c_dec  = (float*)alloc((size_t)Beff * H * 4);
    float* gates  = (float*)alloc((size_t)Beff * G * 4);
    float* hy_pre = (float*)alloc((size_t)Beff * G * 4);
    float* ebias  = (float*)alloc((size_t)G * 4);
    float* dbias  = (float*)alloc((size_t)G * 4);

    // ---- launch helpers ----
    auto cvt = [&](const float* src, _Float16* dst, int Nr, int K, int Kp,
                   int ld, int off) {
        int n = Nr * Kp;
        cvt_pack_kernel<<<(n + 255) / 256, 256, 0, stream>>>(src, dst, Nr, K, Kp, ld, off);
    };
    auto gemm = [&](const _Float16* A1, int lda1, const _Float16* B1, int K1p,
                    const _Float16* A2, int lda2, const _Float16* B2, int K2p,
                    const float* C0, int ldc0, const float* bias,
                    float* Cp, int ldc, _Float16* C16, int ldc16,
                    int Mrows, int Ncols, int act) {
        dim3 grid((Ncols + BN - 1) / BN, (Mrows + BM - 1) / BM);
        wmma_gemm_kernel<<<grid, 256, 0, stream>>>(
            A1, lda1, B1, K1p, A2, lda2, B2, K2p,
            C0, ldc0, bias, Cp, ldc, C16, ldc16, Mrows, Ncols, act);
    };
    auto fill32 = [&](float* p, size_t n) {
        fill0_f32_kernel<<<(unsigned)((n + 255) / 256), 256, 0, stream>>>(p, n);
    };
    auto fill16 = [&](_Float16* p, size_t n) {
        fill0_f16_kernel<<<(unsigned)((n + 255) / 256), 256, 0, stream>>>(p, n);
    };

    // ---- per-launch setup: convert weights, pack x, combine biases, zero state
    cvt(enc_Wih, encWih16,  G, F,  Fp, F,     0);
    cvt(enc_Whh, encWhh16,  G, H,  H,  H,     0);
    cvt(mu_W,    muW16,     Z, H,  H,  H,     0);
    cvt(lv_W,    lvW16,     Z, H,  H,  H,     0);
    cvt(zdec_W,  zdec16,    H, Z,  Z,  Z,     0);
    cvt(dec_Wih, decWihH16, G, H,  H,  F + H, 0);
    cvt(dec_Wih, decWihY16, G, F,  Fp, F + H, H);
    cvt(dec_Whh, decWhh16,  G, H,  H,  H,     0);
    cvt(mlp1_W,  mlp116,    Mm, H, H,  H,     0);
    cvt(mlp2_W,  mlp216,    Mm, Mm, Mm, Mm,   0);
    cvt(out_W,   outW16,    F, Mm, Mm, Mm,    0);
    cvt(cr1_W,   cr116,     H, F,  Fp, F,     0);
    cvt(cr2_W,   cr216,     C3D, H, H, H,     0);

    // pack x: per-step slab t holds le[:, t*F .. t*F+F) of the (Beff, Wt*F) view
    for (int t = 0; t < Wt; ++t)
        cvt(le, xpack + (size_t)t * Beff * Fp, Beff, F, Fp, Wt * F, t * F);

    addvec_kernel<<<(G + 255) / 256, 256, 0, stream>>>(enc_bih, enc_bhh, ebias, G);
    addvec_kernel<<<(G + 255) / 256, 256, 0, stream>>>(dec_bih, dec_bhh, dbias, G);
    fill32(c_enc, (size_t)Beff * H);
    fill32(c_dec, (size_t)Beff * H);
    fill16(hpackE, (size_t)Beff * H);
    fill16(hpackD, (size_t)Beff * H);
    fill16(ypack,  (size_t)Beff * Fp);   // stays zero in padded cols forever

    const int cellBlocks = (Beff * H + 255) / 256;

    // ---- encoder: 50 steps ----
    for (int t = 0; t < Wt; ++t) {
        gemm(xpack + (size_t)t * Beff * Fp, Fp, encWih16, Fp,
             hpackE, H, encWhh16, H,
             nullptr, 0, ebias, gates, G, nullptr, 0, Beff, G, 0);
        lstm_cell_kernel<<<cellBlocks, 256, 0, stream>>>(gates, c_enc, hpackE, Beff, H);
    }

    // ---- VAE heads ----
    gemm(hpackE, H, muW16, H, nullptr, 0, nullptr, 0,
         nullptr, 0, mu_b, mu_out, Z, mupack, Z, Beff, Z, 0);
    gemm(hpackE, H, lvW16, H, nullptr, 0, nullptr, 0,
         nullptr, 0, lv_b, lv_out, Z, nullptr, 0, Beff, Z, 0);
    // z = mu (eval mode); h_y = z @ zdec^T + b  (consumed only as f16)
    gemm(mupack, Z, zdec16, Z, nullptr, 0, nullptr, 0,
         nullptr, 0, zdec_b, nullptr, 0, hypack, H, Beff, H, 0);
    // hy_pre = h_y @ dec_Wih[:, :H]^T + (dec_bih + dec_bhh)  (constant over steps)
    gemm(hypack, H, decWihH16, H, nullptr, 0, nullptr, 0,
         nullptr, 0, dbias, hy_pre, G, nullptr, 0, Beff, G, 0);

    // ---- decoder: 50 steps; y_t stored strided into d_out + packed for t+1 ----
    for (int t = 0; t < Wt; ++t) {
        gemm(ypack, Fp, decWihY16, Fp,
             hpackD, H, decWhh16, H,
             hy_pre, G, nullptr, gates, G, nullptr, 0, Beff, G, 0);
        lstm_cell_kernel<<<cellBlocks, 256, 0, stream>>>(gates, c_dec, hpackD, Beff, H);
        gemm(hpackD, H, mlp116, H, nullptr, 0, nullptr, 0,
             nullptr, 0, mlp1_b, nullptr, 0, m1pack, Mm, Beff, Mm, 1);
        gemm(m1pack, Mm, mlp216, Mm, nullptr, 0, nullptr, 0,
             nullptr, 0, mlp2_b, nullptr, 0, m2pack, Mm, Beff, Mm, 1);
        // y_t -> pred rows (b_eff*Wt + t): f32 ptr = pred + t*F (ldc = Wt*F),
        // plus f16 copy into ypack (padded cols untouched -> remain zero)
        gemm(m2pack, Mm, outW16, Mm, nullptr, 0, nullptr, 0,
             nullptr, 0, out_b, pred + (size_t)t * F, Wt * F, ypack, Fp, Beff, F, 0);
    }

    // ---- coeff_3dmm = tanh(pred @ cr1^T + b1) @ cr2^T + b2 ----
    cvt(pred, predpack, Bb * T, F, Fp, F, 0);     // pack all 96000 frames
    const int rows = Bb * T;                      // 96000 = 15 * CH
    for (int r0 = 0; r0 < rows; r0 += CH) {
        gemm(predpack + (size_t)r0 * Fp, Fp, cr116, Fp, nullptr, 0, nullptr, 0,
             nullptr, 0, cr1_b, nullptr, 0, tmpcpack, H, CH, H, 1);
        gemm(tmpcpack, H, cr216, H, nullptr, 0, nullptr, 0,
             nullptr, 0, cr2_b, coeff + (size_t)r0 * C3D, C3D, nullptr, 0, CH, C3D, 0);
    }
}
